// INRF_32332513804468
// MI455X (gfx1250) — matrix operand order repaired and checked by reading, not compiled, not run on hardware
//
#include <hip/hip_runtime.h>

typedef __attribute__((ext_vector_type(16))) _Float16 v16h;
typedef __attribute__((ext_vector_type(8)))  _Float16 v8h;
typedef __attribute__((ext_vector_type(8)))  float    v8f;

#define HW    1024   // H*W
#define KREAL 144    // KH*KW*C
#define KPAD  160    // padded to 5 K-tiles of 32
#define NROWS 4096   // B*H*W
#define IJPB  2      // output pixels per block (amortizes A/X L2 traffic)

// Kernel 1: im2col X (B,H,W,C) f32 -> A[4096][160] f16 (zero-padded halo + K-pad)
__global__ __launch_bounds__(256) void inrf_im2col(const float* __restrict__ X,
                                                   _Float16* __restrict__ A) {
  int idx = blockIdx.x * 256 + threadIdx.x;
  if (idx >= NROWS * KPAD) return;
  int row = idx / KPAD;
  int col = idx - row * KPAD;
  _Float16 v = (_Float16)0.f;
  if (col < KREAL) {
    int b = row >> 10, pq = row & 1023, p = pq >> 5, q = pq & 31;
    int kk = col >> 4, c = col & 15;
    int kh = kk / 3, kw = kk - kh * 3;
    int ip = p + kh - 1, iq = q + kw - 1;
    if ((unsigned)ip < 32u && (unsigned)iq < 32u)
      v = (_Float16)X[(((b << 10) + (ip << 5) + iq) << 4) + c];
  }
  A[idx] = v;
}

// Kernel 2: one block per IJPB output pixels. Fused conv(WMMA) + relu-diff +
// weighted reductions; each A/X load feeds IJPB WMMA chains; M2/W2 weight rows
// are wave-uniform -> scalar loads; half-select done on VALUES at constant
// indices (single cndmask each, no dynamic register-array indexing).
__global__ __launch_bounds__(256) void inrf_main(const float* __restrict__ X,
                                                 const float* __restrict__ M2,
                                                 const float* __restrict__ W2,
                                                 const float* __restrict__ G,
                                                 const _Float16* __restrict__ A,
                                                 float* __restrict__ Out) {
  __shared__ float part[IJPB * 4 * 8 * 16];  // [jj][b][wave][f]
  const int ij0  = blockIdx.x * IJPB;
  const int tid  = threadIdx.x;
  const int lane = tid & 31;
  const int half = lane >> 4;      // 0: lanes 0-15, 1: lanes 16-31
  const bool hi  = (half != 0);    // loop-invariant divergent predicate
  const int lf   = lane & 15;      // N index (f) / M row-within-tile selector
  // wave index, provably wave-uniform (enables s_load of weight rows)
  const int wu   = __builtin_amdgcn_readfirstlane(tid >> 5);

  // ---- Load B fragments: G_ij viewed as K x N (160 x 16), f16 ----
  // B k order used here (contiguous, matching A): lane half 0 holds K = h, half 1 holds K = 16+h,
  // N = lane % 16.  G_ij rows are 16 contiguous f32 -> coalesced loads.
  v16h bf[IJPB][5];
#pragma unroll
  for (int jj = 0; jj < IJPB; ++jj) {
    const float* gij = G + (size_t)(ij0 + jj) * (KREAL * 16);
#pragma unroll
    for (int kt = 0; kt < 5; ++kt) {
      v16h bv;
#pragma unroll
      for (int h = 0; h < 16; ++h) {
        int K = kt * 32 + half * 16 + h;
        float gv = (K < KREAL) ? gij[K * 16 + lf] : 0.f;
        bv[h] = (_Float16)gv;
      }
      bf[jj][kt] = bv;
    }
  }

  const float* m2base[IJPB];
  const float* w2base[IJPB];
#pragma unroll
  for (int jj = 0; jj < IJPB; ++jj) {
    m2base[jj] = M2 + (size_t)(ij0 + jj) * HW;
    w2base[jj] = W2 + (size_t)(ij0 + jj) * HW;
  }

  // ---- Main loop: 4 batches x 64 row-tiles (8 tiles per wave) ----
  for (int bb = 0; bb < 4; ++bb) {
    float acc[IJPB];
#pragma unroll
    for (int jj = 0; jj < IJPB; ++jj) acc[jj] = 0.f;

    // Hoisted, strength-reduced pointers (advance by constant per tile).
    const _Float16* arow = A + (size_t)((bb << 10) + (wu << 4) + lf) * KPAD;
    const float*    xb   = X + ((size_t)((bb << 10) + (wu << 4)) << 4);
    int pqbase = wu << 4;  // wave-uniform

    for (int tt = 0; tt < 8; ++tt) {
      v8f c[IJPB];
#pragma unroll
      for (int jj = 0; jj < IJPB; ++jj) c[jj] = (v8f){};

#pragma unroll
      for (int kt = 0; kt < 5; ++kt) {
        // A k order used here (contiguous, same as B): half 0 -> K [0..15];
        // half 1 -> [16..31].  Two 16B loads, shared by IJPB chains.
        const v8h lo = *(const v8h*)(arow + kt * 32 + half * 16);
        const v8h hv = *(const v8h*)(arow + kt * 32 + half * 16 + 8);
        v16h a;
#pragma unroll
        for (int i = 0; i < 8; ++i) { a[i] = lo[i]; a[8 + i] = hv[i]; }
#pragma unroll
        for (int jj = 0; jj < IJPB; ++jj)
          c[jj] = __builtin_amdgcn_wmma_f32_16x16x32_f16(false, a, false, bf[jj][kt],
                                                         (short)0, c[jj], false, false);
      }

      // Wave-uniform weight rows -> scalar loads at constant offsets; select
      // the half-dependent value with one cndmask each (constant indices only).
      float wm8[IJPB][8], ww8[IJPB][8];
#pragma unroll
      for (int jj = 0; jj < IJPB; ++jj) {
        const float* mrow = m2base[jj] + pqbase;
        const float* wrow = w2base[jj] + pqbase;
#pragma unroll
        for (int i = 0; i < 8; ++i) {
          float mlo = mrow[i], mhi = mrow[i + 8];
          float wlo = wrow[i], whi = wrow[i + 8];
          wm8[jj][i] = hi ? mhi : mlo;
          ww8[jj][i] = hi ? whi : wlo;
        }
      }

      // Fused epilogue: out_term = M2*x - W2*relu(x - shifted), reduced over pq.
#pragma unroll
      for (int vr = 0; vr < 8; ++vr) {
        // C/D layout: M = vr + 8*half
        float xv = xb[(vr << 4) + (half << 7) + lf];
#pragma unroll
        for (int jj = 0; jj < IJPB; ++jj) {
          float d = xv - c[jj][vr];
          d = d > 0.f ? d : 0.f;
          acc[jj] += wm8[jj][vr] * xv - ww8[jj][vr] * d;
        }
      }

      arow   += (size_t)128 * KPAD;  // 8 tiles * 16 rows
      xb     += (size_t)128 * 16;
      pqbase += 128;
    }

    // Combine the two half-lanes holding the same f, stash per-wave partials.
#pragma unroll
    for (int jj = 0; jj < IJPB; ++jj) {
      float a2 = acc[jj] + __shfl_xor(acc[jj], 16);
      if (half == 0) part[((jj * 4 + bb) * 8 + wu) * 16 + lf] = a2;
    }
  }

  __syncthreads();
  // Deterministic fixed-order cross-wave reduction; IJPB*64 outputs per block.
  if (tid < IJPB * 64) {
    int jj = tid >> 6, b = (tid >> 4) & 3, f = tid & 15;
    float s = 0.f;
#pragma unroll
    for (int w = 0; w < 8; ++w) s += part[((jj * 4 + b) * 8 + w) * 16 + f];
    Out[(((size_t)(b << 10) + (ij0 + jj)) << 4) + f] = s;
  }
}

extern "C" void kernel_launch(void* const* d_in, const int* in_sizes, int n_in,
                              void* d_out, int out_size, void* d_ws, size_t ws_size,
                              hipStream_t stream) {
  (void)in_sizes; (void)n_in; (void)out_size; (void)ws_size;
  const float* X  = (const float*)d_in[0];  // inputs (4,32,32,16)
  const float* M2 = (const float*)d_in[1];  // M  -> (1024,1024)
  const float* W2 = (const float*)d_in[2];  // Wp -> (1024,1024)
  const float* G  = (const float*)d_in[3];  // G  (1024,144,16)
  float* Out = (float*)d_out;
  _Float16* A = (_Float16*)d_ws;            // 4096*160*2 = 1.25 MB scratch

  inrf_im2col<<<(NROWS * KPAD + 255) / 256, 256, 0, stream>>>(X, A);
  inrf_main<<<HW / IJPB, 256, 0, stream>>>(X, M2, W2, G, A, Out);
}
